// RSSM_18674517803087
// MI455X (gfx1250) — matrix operand order repaired and checked by reading, not compiled, not run on hardware
//
#include <hip/hip_runtime.h>
#include <hip/hip_bf16.h>
#include <math.h>

// ---------------------------------------------------------------------------
// RSSM (DreamerV3-style) forward for MI455X / gfx1250, wave32 + WMMA.
// GEMMs: v_wmma_f32_16x16x32_bf16, 128x64 block tile, 8 waves, each wave
// computing a 32x32 tile (2x2 accumulators, 4 WMMAs per K-step, fragment
// reuse). Weights pre-converted to bf16 and pre-transposed to [N][K]; all
// staging is b128 global load + b128 LDS store, register double-buffered.
// Fragment build is pure aligned ds_load_b128 (zero VALU).
// ---------------------------------------------------------------------------

typedef __attribute__((ext_vector_type(16))) __bf16 v16bf;
typedef __attribute__((ext_vector_type(8)))  float  v8f;

// Model dims
constexpr int kB   = 32;
constexpr int kT   = 128;
constexpr int kNT  = kB * kT;     // 4096 tokens
constexpr int kOBS = 512;
constexpr int kACT = 64;
constexpr int kD   = 512;
constexpr int kDI  = 1024;        // 2*D
constexpr int kDS  = 16;
constexpr int kDTR = 32;
constexpr int kLAT = 1024;        // 32*32
constexpr int kNE  = 5;

__device__ __forceinline__ unsigned short f2bfu(float f) {
  union { float f; unsigned u; } in; in.f = f;
  unsigned r = in.u + 0x7FFFu + ((in.u >> 16) & 1u);   // round-to-nearest-even
  return (unsigned short)(r >> 16);
}

__device__ __forceinline__ float siluf(float x) { return x / (1.0f + __expf(-x)); }

// Vectorized f32 -> bf16 conversion (4 elements / thread), same layout.
__global__ __launch_bounds__(256)
void f32_to_bf16_kernel(const float* __restrict__ src, unsigned short* __restrict__ dst, int n4)
{
  int i = blockIdx.x * blockDim.x + threadIdx.x;
  if (i >= n4) return;
  float4 v = ((const float4*)src)[i];
  uint2 o;
  o.x = (unsigned)f2bfu(v.x) | ((unsigned)f2bfu(v.y) << 16);
  o.y = (unsigned)f2bfu(v.z) | ((unsigned)f2bfu(v.w) << 16);
  ((uint2*)dst)[i] = o;
}

// f32 [K][N] -> bf16 [N][K] (transpose + convert), LDS-tiled 32x32.
// Grid: (N/32, K/32), block 256 (32x8).
__global__ __launch_bounds__(256)
void cvt_transpose_kernel(const float* __restrict__ src, unsigned short* __restrict__ dst,
                          int K, int N)
{
  __shared__ float tile[32][33];
  int tk0 = blockIdx.y * 32;
  int tn0 = blockIdx.x * 32;
  int tx = threadIdx.x & 31, ty = threadIdx.x >> 5;
  #pragma unroll
  for (int j = 0; j < 32; j += 8)
    tile[ty + j][tx] = src[(size_t)(tk0 + ty + j) * N + tn0 + tx];
  __syncthreads();
  #pragma unroll
  for (int j = 0; j < 32; j += 8)
    dst[(size_t)(tn0 + ty + j) * K + tk0 + tx] = f2bfu(tile[tx][ty + j]);
}

// ---------------------------------------------------------------------------
// Tiled WMMA GEMM:  out = epilogue(A[M,K](lda,bf16) @ Wt[N,K](bf16)^T + bias)
// act: 0=none, 1=silu, 2=softplus.  Optional addsrc (residual), optional
// f32 C / bf16 Cbf outputs.  M%128==0, N%64==0, K%32==0, lda%8==0.
// Block 256 (8 waves), tile 128x64, BK=32; each wave owns 32x32 (2x2 accums).
// ---------------------------------------------------------------------------
__global__ __launch_bounds__(256)
void gemm_bf16_wmma_kernel(const unsigned short* __restrict__ A,
                           const unsigned short* __restrict__ Wt,
                           const float* __restrict__ bias, const float* __restrict__ addsrc,
                           float* __restrict__ C, unsigned short* __restrict__ Cbf,
                           int M, int N, int K, int lda, int act)
{
  __shared__ alignas(32) unsigned short As[128 * 48];  // row stride 48 bf16 (96B)
  __shared__ alignas(32) unsigned short Bs[64 * 48];   // col-major: Bs[n][k]

  const int tid   = threadIdx.x;
  const int lane  = tid & 31;
  const int wave  = tid >> 5;
  const int wm    = wave & 3;      // M sub-tile (32-row granulate)
  const int wn    = wave >> 2;     // N sub-tile (32-col)
  const int m0    = blockIdx.y * 128;
  const int n0    = blockIdx.x * 64;
  const int m16   = lane & 15;
  const int khalf = lane >> 4;

  // Staging map: thread -> (row 0..63, 8 contiguous K); A needs rows 0..127.
  const int ar = tid >> 2;
  const int ak = (tid & 3) << 3;                       // 0,8,16,24
  const int pa = ak;                                   // A contiguous k order (matches B)
  const unsigned short* Ap0 = A  + (size_t)(m0 + ar) * lda + ak;
  const unsigned short* Ap1 = A  + (size_t)(m0 + 64 + ar) * lda + ak;
  const unsigned short* Wp  = Wt + (size_t)(n0 + ar) * K   + ak;
  unsigned short* asDst0 = As + ar * 48 + pa;
  unsigned short* asDst1 = As + (64 + ar) * 48 + pa;
  unsigned short* bsDst  = Bs + ar * 48 + ak;

  v8f acc00 = {0.f,0.f,0.f,0.f,0.f,0.f,0.f,0.f};
  v8f acc01 = acc00, acc10 = acc00, acc11 = acc00;

  uint4 a0Reg = *(const uint4*)Ap0;                    // first tile
  uint4 a1Reg = *(const uint4*)Ap1;
  uint4 bReg  = *(const uint4*)Wp;

  for (int k0 = 0;;) {
    *(uint4*)asDst0 = a0Reg;                           // ds_store_b128 x3
    *(uint4*)asDst1 = a1Reg;
    *(uint4*)bsDst  = bReg;
    int kn = k0 + 32;
    bool more = kn < K;
    if (more) {                                        // issue next-tile loads early
      a0Reg = *(const uint4*)(Ap0 + kn);
      a1Reg = *(const uint4*)(Ap1 + kn);
      bReg  = *(const uint4*)(Wp + kn);
      if (kn + 32 < K) {                               // prefetch tile after next
        __builtin_prefetch(Ap0 + kn + 32, 0, 1);
        __builtin_prefetch(Ap1 + kn + 32, 0, 1);
        __builtin_prefetch(Wp + kn + 32, 0, 1);
      }
    }
    __syncthreads();

    v16bf a0 = *(const v16bf*)(As + (wm * 32 + m16) * 48 + khalf * 16);
    v16bf a1 = *(const v16bf*)(As + (wm * 32 + 16 + m16) * 48 + khalf * 16);
    v16bf b0 = *(const v16bf*)(Bs + (wn * 32 + m16) * 48 + khalf * 16);
    v16bf b1 = *(const v16bf*)(Bs + (wn * 32 + 16 + m16) * 48 + khalf * 16);
    acc00 = __builtin_amdgcn_wmma_f32_16x16x32_bf16(false, a0, false, b0, (short)0, acc00, false, false);
    acc01 = __builtin_amdgcn_wmma_f32_16x16x32_bf16(false, a0, false, b1, (short)0, acc01, false, false);
    acc10 = __builtin_amdgcn_wmma_f32_16x16x32_bf16(false, a1, false, b0, (short)0, acc10, false, false);
    acc11 = __builtin_amdgcn_wmma_f32_16x16x32_bf16(false, a1, false, b1, (short)0, acc11, false, false);

    if (!more) break;
    k0 = kn;
    __syncthreads();
  }

  // Epilogue: C layout VGPR r -> row khalf*8 + r, col = lane&15.
  #pragma unroll
  for (int r = 0; r < 8; ++r) {
    int rowA = m0 + wm * 32 + khalf * 8 + r;
    int rowB = rowA + 16;
    int c0   = n0 + wn * 32 + m16;
    int c1   = c0 + 16;
    float vals[4] = { acc00[r], acc01[r], acc10[r], acc11[r] };
    int   rows[4] = { rowA, rowA, rowB, rowB };
    int   cols[4] = { c0, c1, c0, c1 };
    #pragma unroll
    for (int q = 0; q < 4; ++q) {
      float v = vals[q];
      if (bias) v += bias[cols[q]];
      if (act == 1)      v = siluf(v);
      else if (act == 2) v = (v > 20.f) ? v : log1pf(__expf(v));
      size_t off = (size_t)rows[q] * N + cols[q];
      if (addsrc) v += addsrc[off];
      if (C)   C[off] = v;
      if (Cbf) Cbf[off] = f2bfu(v);
    }
  }
}

// ---------------------------------------------------------------------------
// Softmax + unimix (+ optional gumbel straight-through one-hot sample).
// One wave per 32-class row.  z forward value == one-hot exactly.
// ---------------------------------------------------------------------------
__global__ __launch_bounds__(256)
void cat_head_kernel(const float* __restrict__ logits, const float* __restrict__ gumbel,
                     float* __restrict__ probs_out, float* __restrict__ z_out, int nrows)
{
  int row  = blockIdx.x * 8 + (threadIdx.x >> 5);
  int lane = threadIdx.x & 31;
  if (row >= nrows) return;
  float l = logits[(size_t)row * 32 + lane];
  float m = l;
  for (int off = 16; off; off >>= 1) m = fmaxf(m, __shfl_xor(m, off, 32));
  float e = __expf(l - m);
  float s = e;
  for (int off = 16; off; off >>= 1) s += __shfl_xor(s, off, 32);
  float p = 0.99f * (e / s) + 0.01f / 32.0f;
  probs_out[(size_t)row * 32 + lane] = p;
  if (z_out) {
    float score = logf(p) + gumbel[(size_t)row * 32 + lane];
    int idx = lane;
    for (int off = 16; off; off >>= 1) {
      float so = __shfl_xor(score, off, 32);
      int   io = __shfl_xor(idx,   off, 32);
      if (so > score || (so == score && io < idx)) { score = so; idx = io; }
    }
    z_out[(size_t)row * 32 + lane] = (lane == idx) ? 1.0f : 0.0f;
  }
}

// RMSNorm over D=512: one wave per row; optional f32 and bf16 outputs.
__global__ __launch_bounds__(256)
void rmsnorm_kernel(const float* __restrict__ x, const float* __restrict__ w,
                    float* __restrict__ fout, unsigned short* __restrict__ bfout, int nrows)
{
  int row  = blockIdx.x * 8 + (threadIdx.x >> 5);
  int lane = threadIdx.x & 31;
  if (row >= nrows) return;
  const float* xr = x + (size_t)row * kD;
  float v[16]; float ss = 0.f;
  #pragma unroll
  for (int i = 0; i < 16; ++i) { v[i] = xr[i * 32 + lane]; ss += v[i] * v[i]; }
  for (int off = 16; off; off >>= 1) ss += __shfl_xor(ss, off, 32);
  float rs = rsqrtf(ss * (1.0f / kD) + 1e-5f);
  #pragma unroll
  for (int i = 0; i < 16; ++i) {
    float o = v[i] * rs * w[i * 32 + lane];
    if (fout)  fout[(size_t)row * kD + i * 32 + lane] = o;
    if (bfout) bfout[(size_t)row * kD + i * 32 + lane] = f2bfu(o);
  }
}

// Concat [z (1024) | a (64)] -> bf16 za (1088 cols); only consumed by a GEMM.
__global__ void concat_za_kernel(const float* __restrict__ z, const float* __restrict__ a,
                                 unsigned short* __restrict__ za)
{
  int idx = blockIdx.x * blockDim.x + threadIdx.x;
  if (idx >= kNT * (kLAT + kACT)) return;
  int row = idx / (kLAT + kACT), col = idx % (kLAT + kACT);
  float v = (col < kLAT) ? z[(size_t)row * kLAT + col]
                         : a[(size_t)row * kACT + (col - kLAT)];
  za[idx] = f2bfu(v);
}

// Depthwise causal conv (K=4, left pad 3) over xm half of xz, + bias, silu.
// Writes f32 (for scan) and bf16 (for xproj GEMM).
__global__ void conv_silu_kernel(const float* __restrict__ xz, const float* __restrict__ cw,
                                 const float* __restrict__ cb,
                                 float* __restrict__ xm, unsigned short* __restrict__ xm_bf)
{
  int idx = blockIdx.x * blockDim.x + threadIdx.x;      // over B*T*DI
  if (idx >= kB * kT * kDI) return;
  int d = idx % kDI;
  int t = (idx / kDI) % kT;
  int b = idx / (kDI * kT);
  float s = cb[d];
  #pragma unroll
  for (int j = 0; j < 4; ++j) {
    int tt = t - 3 + j;
    if (tt >= 0) s += xz[((size_t)(b * kT + tt)) * (2 * kDI) + d] * cw[d * 4 + j];
  }
  s = siluf(s);
  xm[idx] = s;
  xm_bf[idx] = f2bfu(s);
}

// Selective scan: one thread per (b, d); 16-state in registers, T=128 steps.
// y = (scan + x*Dd) * silu(z); emitted as bf16 (only consumed by out-proj GEMM).
__global__ __launch_bounds__(256)
void scan_kernel(const float* __restrict__ xm, const float* __restrict__ dt,
                 const float* __restrict__ xdbl, const float* __restrict__ xz,
                 const float* __restrict__ A_log, const float* __restrict__ Dd,
                 unsigned short* __restrict__ y_bf)
{
  int idx = blockIdx.x * blockDim.x + threadIdx.x;      // B*DI = 32768
  if (idx >= kB * kDI) return;
  int d = idx & (kDI - 1);
  int b = idx >> 10;
  float Arow[kDS], h[kDS];
  #pragma unroll
  for (int s = 0; s < kDS; ++s) { Arow[s] = -__expf(A_log[d * kDS + s]); h[s] = 0.f; }
  float Ddv = Dd[d];
  for (int t = 0; t < kT; ++t) {
    size_t rbt = (size_t)(b * kT + t);
    float x   = xm[rbt * kDI + d];
    float dtv = dt[rbt * kDI + d];
    const float* xd = xdbl + rbt * 64;
    float dx = dtv * x, acc = 0.f;
    #pragma unroll
    for (int s = 0; s < kDS; ++s) {
      float dA = __expf(dtv * Arow[s]);
      h[s] = dA * h[s] + dx * xd[32 + s];    // B_t
      acc += h[s] * xd[48 + s];              // C_t
    }
    acc += x * Ddv;
    float zv = xz[rbt * (2 * kDI) + kDI + d];
    acc *= siluf(zv);
    y_bf[rbt * kDI + d] = f2bfu(acc);
  }
}

// GEMV over D=512: out[n] = X[n,:] . w + b[0]. One wave per row.
__global__ __launch_bounds__(256)
void gemv512_kernel(const float* __restrict__ X, const float* __restrict__ w,
                    const float* __restrict__ bptr, float* __restrict__ out, int nrows)
{
  int row  = blockIdx.x * 8 + (threadIdx.x >> 5);
  int lane = threadIdx.x & 31;
  if (row >= nrows) return;
  float s = 0.f;
  #pragma unroll
  for (int i = 0; i < 16; ++i) s += X[(size_t)row * kD + i * 32 + lane] * w[i * 32 + lane];
  for (int off = 16; off; off >>= 1) s += __shfl_xor(s, off, 32);
  if (lane == 0) out[row] = s + bptr[0];
}

// Ensemble mean / std (ddof=1) over NE=5 members: r is [NE][n].
__global__ void meanstd_kernel(const float* __restrict__ r, float* __restrict__ mean_out,
                               float* __restrict__ std_out, int n)
{
  int i = blockIdx.x * blockDim.x + threadIdx.x;
  if (i >= n) return;
  float m = 0.f;
  #pragma unroll
  for (int e = 0; e < kNE; ++e) m += r[(size_t)e * n + i];
  m *= (1.0f / kNE);
  float v = 0.f;
  #pragma unroll
  for (int e = 0; e < kNE; ++e) { float d = r[(size_t)e * n + i] - m; v += d * d; }
  mean_out[i] = m;
  std_out[i]  = sqrtf(v / (kNE - 1));
}

// ---------------------------------------------------------------------------
// Host orchestration
// ---------------------------------------------------------------------------
static inline void launch_gemm(const unsigned short* A, const unsigned short* Wt,
                               const float* bias, const float* addsrc,
                               float* C, unsigned short* Cbf,
                               int M, int N, int K, int lda, int act, hipStream_t s)
{
  dim3 grid(N / 64, M / 128);
  gemm_bf16_wmma_kernel<<<grid, 256, 0, s>>>(A, Wt, bias, addsrc, C, Cbf, M, N, K, lda, act);
}

static inline void launch_cvt(const float* src, unsigned short* dst, size_t n, hipStream_t s)
{
  int n4 = (int)(n / 4);
  f32_to_bf16_kernel<<<(n4 + 255) / 256, 256, 0, s>>>(src, dst, n4);
}

static inline void launch_cvt_tr(const float* src, unsigned short* dst, int K, int N, hipStream_t s)
{
  dim3 grid(N / 32, K / 32);
  cvt_transpose_kernel<<<grid, 256, 0, s>>>(src, dst, K, N);
}

extern "C" void kernel_launch(void* const* d_in, const int* in_sizes, int n_in,
                              void* d_out, int out_size, void* d_ws, size_t ws_size,
                              hipStream_t stream) {
  (void)in_sizes; (void)n_in; (void)out_size; (void)ws_size;
  auto F = [&](int i) { return (const float*)d_in[i]; };

  // Inputs in setup_inputs() dict insertion order.
  const float* x_BT   = F(0);
  const float* a_BT   = F(1);
  const float* gumbel = F(2);
  const float* enc_b1 = F(4);
  const float* enc_b2 = F(6);
  const float* post_b = F(8);
  const float* inp_b  = F(10);
  const float* fnorm_w = F(11);
  const float* prior_b1 = F(13);
  const float* prior_b2 = F(15);
  const float* rew_b1 = F(17);
  const float* rew_w2 = F(18); const float* rew_b2 = F(19);
  const float* cont_b1 = F(21);
  const float* cont_w2 = F(22); const float* cont_b2 = F(23);
  const int LBASE = 24, LSTRIDE = 12;   // norm_w,in_w,in_b,conv_w,conv_b,xproj_w,dt_w,dt_b,A_log,Dd,out_w,out_b

  // Output layout (flat, return order)
  float* out       = (float*)d_out;
  float* z_out     = out;                                   // B*T*1024
  float* hbt_out   = out + (size_t)kNT * kLAT;              // B*T*512
  float* post_out  = hbt_out + (size_t)kNT * kD;            // B*T*1024
  float* prior_out = post_out + (size_t)kNT * kLAT;         // B*T*1024
  float* rmean_out = prior_out + (size_t)kNT * kLAT;        // B*T
  float* rstd_out  = rmean_out + kNT;                       // B*T
  float* cont_out  = rstd_out + kNT;                        // B*T

  // ---- Workspace: f32 region then bf16 (ushort) region ----
  float* ws = (float*)d_ws;
  size_t o = 0;
  float* xz   = ws + o;  o += (size_t)kNT * 2 * kDI;   // 8.4M f32
  float* xm   = ws + o;  o += (size_t)kNT * kDI;
  float* dtb  = ws + o;  o += (size_t)kNT * kDI;
  float* hb   = ws + o;  o += (size_t)kNT * kD;
  float* tmp  = ws + o;  o += (size_t)kNT * kD;
  float* xdbl = ws + o;  o += (size_t)kNT * 64;
  float* rbuf = ws + o;  o += (size_t)kNE * kNT;

  unsigned short* bfbase = (unsigned short*)(ws + o);
  size_t u = 0;
  unsigned short* x_bf    = bfbase + u;  u += (size_t)kNT * kOBS;
  unsigned short* tmp_bf  = bfbase + u;  u += (size_t)kNT * kD;
  unsigned short* hn_bf   = bfbase + u;  u += (size_t)kNT * kD;
  unsigned short* hbt_bf  = bfbase + u;  u += (size_t)kNT * kD;
  unsigned short* za_bf   = bfbase + u;  u += (size_t)kNT * (kLAT + kACT);
  unsigned short* xm_bf   = bfbase + u;  u += (size_t)kNT * kDI;
  unsigned short* xdbl_bf = bfbase + u;  u += (size_t)kNT * 64;
  unsigned short* yb_bf   = bfbase + u;  u += (size_t)kNT * kDI;
  // transposed bf16 weight copies ([N][K]), converted once per launch
  unsigned short* enc_w1_bf   = bfbase + u;  u += (size_t)kOBS * kD;
  unsigned short* enc_w2_bf   = bfbase + u;  u += (size_t)kD * kD;
  unsigned short* post_w_bf   = bfbase + u;  u += (size_t)kD * kLAT;
  unsigned short* inp_w_bf    = bfbase + u;  u += (size_t)(kLAT + kACT) * kD;
  unsigned short* prior_w1_bf = bfbase + u;  u += (size_t)kD * kD;
  unsigned short* prior_w2_bf = bfbase + u;  u += (size_t)kD * kLAT;
  unsigned short* cont_w1_bf  = bfbase + u;  u += (size_t)kD * kD;
  unsigned short* rew_w1_bf   = bfbase + u;  u += (size_t)kNE * kD * kD;
  unsigned short* lw_bf[2][4];   // in_w, xproj_w, dt_w, out_w per layer
  for (int L = 0; L < 2; ++L) {
    lw_bf[L][0] = bfbase + u;  u += (size_t)kD * 2 * kDI;
    lw_bf[L][1] = bfbase + u;  u += (size_t)kDI * 64;
    lw_bf[L][2] = bfbase + u;  u += (size_t)kDTR * kDI;
    lw_bf[L][3] = bfbase + u;  u += (size_t)kDI * kD;
  }

  // ---- Convert observation (row-major) + weights (transposed) to bf16 ----
  launch_cvt(x_BT, x_bf, (size_t)kNT * kOBS, stream);
  launch_cvt_tr(F(3),  enc_w1_bf,   kOBS, kD, stream);
  launch_cvt_tr(F(5),  enc_w2_bf,   kD, kD, stream);
  launch_cvt_tr(F(7),  post_w_bf,   kD, kLAT, stream);
  launch_cvt_tr(F(9),  inp_w_bf,    kLAT + kACT, kD, stream);
  launch_cvt_tr(F(12), prior_w1_bf, kD, kD, stream);
  launch_cvt_tr(F(14), prior_w2_bf, kD, kLAT, stream);
  launch_cvt_tr(F(20), cont_w1_bf,  kD, kD, stream);
  for (int e = 0; e < kNE; ++e)
    launch_cvt_tr(F(16) + (size_t)e * kD * kD, rew_w1_bf + (size_t)e * kD * kD, kD, kD, stream);
  for (int L = 0; L < 2; ++L) {
    launch_cvt_tr(F(LBASE + L * LSTRIDE + 1),  lw_bf[L][0], kD, 2 * kDI, stream);
    launch_cvt_tr(F(LBASE + L * LSTRIDE + 5),  lw_bf[L][1], kDI, 64, stream);
    launch_cvt_tr(F(LBASE + L * LSTRIDE + 6),  lw_bf[L][2], kDTR, kDI, stream);
    launch_cvt_tr(F(LBASE + L * LSTRIDE + 10), lw_bf[L][3], kDI, kD, stream);
  }

  const int NROWCAT = kNT * 32;   // 131072 softmax rows

  // 1) Encoder: silu(x@w1+b1) @ w2 + b2  (bf16 chain through tmp_bf / hn_bf)
  launch_gemm(x_bf,   enc_w1_bf, enc_b1, nullptr, nullptr, tmp_bf, kNT, kD, kOBS, kOBS, 1, stream);
  launch_gemm(tmp_bf, enc_w2_bf, enc_b2, nullptr, nullptr, hn_bf,  kNT, kD, kD,   kD,   0, stream);
  // 2) Posterior logits -> probs + straight-through one-hot z
  launch_gemm(hn_bf, post_w_bf, post_b, nullptr, xz, nullptr, kNT, kLAT, kD, kD, 0, stream);
  cat_head_kernel<<<NROWCAT / 8, 256, 0, stream>>>(xz, gumbel, post_out, z_out, NROWCAT);
  // 3) h = concat(z, a) @ inp_w + inp_b
  concat_za_kernel<<<(kNT * (kLAT + kACT) + 255) / 256, 256, 0, stream>>>(z_out, a_BT, za_bf);
  launch_gemm(za_bf, inp_w_bf, inp_b, nullptr, hb, nullptr, kNT, kD, kLAT + kACT, kLAT + kACT, 0, stream);

  // 4) Mamba layers
  for (int L = 0; L < 2; ++L) {
    const float* norm_w = F(LBASE + L * LSTRIDE + 0);
    const float* in_b   = F(LBASE + L * LSTRIDE + 2);
    const float* conv_w = F(LBASE + L * LSTRIDE + 3);
    const float* conv_b = F(LBASE + L * LSTRIDE + 4);
    const float* dt_b   = F(LBASE + L * LSTRIDE + 7);
    const float* A_log  = F(LBASE + L * LSTRIDE + 8);
    const float* Dd     = F(LBASE + L * LSTRIDE + 9);
    const float* out_b  = F(LBASE + L * LSTRIDE + 11);

    rmsnorm_kernel<<<kNT / 8, 256, 0, stream>>>(hb, norm_w, nullptr, hn_bf, kNT);
    launch_gemm(hn_bf, lw_bf[L][0], in_b, nullptr, xz, nullptr, kNT, 2 * kDI, kD, kD, 0, stream);
    conv_silu_kernel<<<(kB * kT * kDI + 255) / 256, 256, 0, stream>>>(xz, conv_w, conv_b, xm, xm_bf);
    launch_gemm(xm_bf, lw_bf[L][1], nullptr, nullptr, xdbl, xdbl_bf, kNT, 64, kDI, kDI, 0, stream);
    // dt = softplus(dt_r @ dt_w + dt_b); dt_r = xdbl[:, :32] (lda=64)
    launch_gemm(xdbl_bf, lw_bf[L][2], dt_b, nullptr, dtb, nullptr, kNT, kDI, kDTR, 64, 2, stream);
    scan_kernel<<<(kB * kDI) / 256, 256, 0, stream>>>(xm, dtb, xdbl, xz, A_log, Dd, yb_bf);
    // h = h + y @ out_w + out_b (residual fused; in-place safe)
    launch_gemm(yb_bf, lw_bf[L][3], out_b, hb, hb, nullptr, kNT, kD, kDI, kDI, 0, stream);
  }

  // 5) Final norm -> f32 h_BT output + bf16 copy for the heads
  rmsnorm_kernel<<<kNT / 8, 256, 0, stream>>>(hb, fnorm_w, hbt_out, hbt_bf, kNT);

  // 6) Prior head
  launch_gemm(hbt_bf, prior_w1_bf, prior_b1, nullptr, nullptr, tmp_bf, kNT, kD, kD, kD, 1, stream);
  launch_gemm(tmp_bf, prior_w2_bf, prior_b2, nullptr, xz, nullptr, kNT, kLAT, kD, kD, 0, stream);
  cat_head_kernel<<<NROWCAT / 8, 256, 0, stream>>>(xz, nullptr, prior_out, nullptr, NROWCAT);

  // 7) Reward ensemble (NE=5)
  for (int e = 0; e < kNE; ++e) {
    launch_gemm(hbt_bf, rew_w1_bf + (size_t)e * kD * kD, rew_b1 + (size_t)e * kD,
                nullptr, tmp, nullptr, kNT, kD, kD, kD, 1, stream);
    gemv512_kernel<<<kNT / 8, 256, 0, stream>>>(tmp, rew_w2 + (size_t)e * kD,
                                                rew_b2 + e, rbuf + (size_t)e * kNT, kNT);
  }
  meanstd_kernel<<<kNT / 256, 256, 0, stream>>>(rbuf, rmean_out, rstd_out, kNT);

  // 8) Continue head
  launch_gemm(hbt_bf, cont_w1_bf, cont_b1, nullptr, tmp, nullptr, kNT, kD, kD, kD, 1, stream);
  gemv512_kernel<<<kNT / 8, 256, 0, stream>>>(tmp, cont_w2, cont_b2, cont_out, kNT);
}
